// GLPE_80358838108897
// MI455X (gfx1250) — matrix operand order repaired and checked by reading, not compiled, not run on hardware
//
#include <hip/hip_runtime.h>

#define NTR   16384
#define OBSD  512
#define HIDD  1024
#define PHID  512
#define NACTD 18
#define KDEPTH 12
#define NINTD 4095
#define NREGD 4096

typedef _Float16 h4   __attribute__((ext_vector_type(4)));
typedef _Float16 h8   __attribute__((ext_vector_type(8)));
typedef _Float16 v16h __attribute__((ext_vector_type(16)));
typedef float    v8f  __attribute__((ext_vector_type(8)));
typedef float    f4   __attribute__((ext_vector_type(4)));

union ABfrag { v16h v; h8 h[2]; };

// ---------------- f32 -> f16 conversion (4-wide) ----------------
__global__ void cvt_f32_f16(const float* __restrict__ in, _Float16* __restrict__ out, int n4) {
  int i = blockIdx.x * blockDim.x + threadIdx.x;
  if (i >= n4) return;
  f4 x = ((const f4*)in)[i];
  h4 y;
  y.x = (_Float16)x.x; y.y = (_Float16)x.y; y.z = (_Float16)x.z; y.w = (_Float16)x.w;
  ((h4*)out)[i] = y;
}

// ------- pack row-major f32 W[K,N] into WMMA-B fragment layout (f16) -------
// Fragment tile (kt,nt): 32 lanes x 16 f16 contiguous (1 KB).
// lane L: column nt*16 + (L&15); element j (0..15): K row kt*32 + 16*(j>>3) + 8*(L>>4) + (j&7).
__global__ void pack_w(const float* __restrict__ W, _Float16* __restrict__ Bp, int K, int N) {
  int idx = blockIdx.x * blockDim.x + threadIdx.x;
  if (idx >= K * N) return;
  int j    = idx & 15;
  int lane = (idx >> 4) & 31;
  int tile = idx >> 9;               // kt * (N/16) + nt
  int ntN  = N >> 4;
  int kt = tile / ntN, nt = tile - kt * ntN;
  int r = kt * 32 + ((j >> 3) << 4) + ((lane >> 4) << 3) + (j & 7);
  int c = (nt << 4) + (lane & 15);
  Bp[idx] = (_Float16)W[(size_t)r * N + c];
}

// ---------------- WMMA GEMM: C = post(A[M,K] @ B[K,N] + bias) ----------------
// A: row-major f16.  Bp: pre-packed fragments.  One wave computes a 16x64
// output tile (1 A fragment reused across 4 B fragments / 4 accumulators per
// K-step), software double-buffered so next-step loads issue under the WMMAs.
// 4 waves/block -> block tile 16x256.  K % 32 == 0, N % 256 == 0, KT >= 2.
template<bool RELU, bool F16OUT, bool F32OUT, bool ACTADD>
__global__ __launch_bounds__(128)
void gemm_wmma_f16(const _Float16* __restrict__ A,
                   const _Float16* __restrict__ Bp,
                   const float* __restrict__ bias,
                   _Float16* __restrict__ C16,
                   float* __restrict__ C32,
                   int M, int N, int K,
                   const float* __restrict__ Wact,   // action-row block base [NACT, N]
                   const int* __restrict__ acts)
{
  const int lane  = threadIdx.x & 31;
  const int wave  = threadIdx.x >> 5;
  const int nt0   = (blockIdx.x * 4 + wave) * 4;    // first of 4 N-tiles
  const int mbase = blockIdx.y << 4;
  const int ntTot = N >> 4;
  const int col   = lane & 15;
  const int hlf   = lane >> 4;

  // A fragment: lane holds row mbase+col; VGPR0-3 = K 8*hlf+0..7, VGPR4-7 = K 16+8*hlf+0..7
  const _Float16* arow = A + (size_t)(mbase + col) * K + (hlf << 3);
  // B fragments for nt0..nt0+3 are contiguous: tile stride = 512 f16 (1 KB)
  const _Float16* bptr = Bp + (((size_t)nt0 * 32) + lane) * 16;
  const size_t bstep = (size_t)ntTot * 512;         // f16 elements per kt step

  auto load_frags = [&](int kt, ABfrag& a, ABfrag& b0, ABfrag& b1,
                        ABfrag& b2, ABfrag& b3) {
    const _Float16* ak = arow + (kt << 5);
    a.h[0] = *(const h8*)(ak);
    a.h[1] = *(const h8*)(ak + 16);
    const _Float16* bk = bptr + (size_t)kt * bstep;
    b0.h[0] = *(const h8*)(bk);          b0.h[1] = *(const h8*)(bk + 8);
    b1.h[0] = *(const h8*)(bk + 512);    b1.h[1] = *(const h8*)(bk + 520);
    b2.h[0] = *(const h8*)(bk + 1024);   b2.h[1] = *(const h8*)(bk + 1032);
    b3.h[0] = *(const h8*)(bk + 1536);   b3.h[1] = *(const h8*)(bk + 1544);
  };

  v8f c0 = {}, c1 = {}, c2 = {}, c3 = {};
  const int KT = K >> 5;

  ABfrag a, b0, b1, b2, b3;
  load_frags(0, a, b0, b1, b2, b3);

  #pragma unroll 2
  for (int kt = 0; kt < KT - 1; ++kt) {
    ABfrag an, b0n, b1n, b2n, b3n;
    load_frags(kt + 1, an, b0n, b1n, b2n, b3n);       // in-flight under WMMAs
    __builtin_prefetch(bptr + (size_t)(kt + 2) * bstep, 0, 1);
    c0 = __builtin_amdgcn_wmma_f32_16x16x32_f16(false, a.v, false, b0.v, (short)0, c0, false, false);
    c1 = __builtin_amdgcn_wmma_f32_16x16x32_f16(false, a.v, false, b1.v, (short)0, c1, false, false);
    c2 = __builtin_amdgcn_wmma_f32_16x16x32_f16(false, a.v, false, b2.v, (short)0, c2, false, false);
    c3 = __builtin_amdgcn_wmma_f32_16x16x32_f16(false, a.v, false, b3.v, (short)0, c3, false, false);
    a = an; b0 = b0n; b1 = b1n; b2 = b2n; b3 = b3n;
  }
  // peeled final K-step
  c0 = __builtin_amdgcn_wmma_f32_16x16x32_f16(false, a.v, false, b0.v, (short)0, c0, false, false);
  c1 = __builtin_amdgcn_wmma_f32_16x16x32_f16(false, a.v, false, b1.v, (short)0, c1, false, false);
  c2 = __builtin_amdgcn_wmma_f32_16x16x32_f16(false, a.v, false, b2.v, (short)0, c2, false, false);
  c3 = __builtin_amdgcn_wmma_f32_16x16x32_f16(false, a.v, false, b3.v, (short)0, c3, false, false);

  v8f acc[4] = { c0, c1, c2, c3 };
  #pragma unroll
  for (int i = 0; i < 8; ++i) {
    const int row = mbase + i + (hlf << 3);
    const size_t rbase = (size_t)row * N;
    int arow_i = 0;
    if (ACTADD) arow_i = acts[row];
    #pragma unroll
    for (int t = 0; t < 4; ++t) {
      const int ncol = ((nt0 + t) << 4) + col;
      float v = acc[t][i] + bias[ncol];
      if (ACTADD) v += Wact[(size_t)arow_i * N + ncol];
      if (RELU)   v = v > 0.f ? v : 0.f;
      if (F32OUT) C32[rbase + ncol] = v;
      if (F16OUT) C16[rbase + ncol] = (_Float16)v;
    }
  }
}

// ---- per-row impact = ||phi1-phi0||, err = mean((pred-phi1)^2); 1 wave/row ----
__global__ __launch_bounds__(256)
void rowstats(const float* __restrict__ phi0, const float* __restrict__ phi1,
              const float* __restrict__ pred, float* __restrict__ impact,
              float* __restrict__ err)
{
  int row  = blockIdx.x * 8 + (threadIdx.x >> 5);
  int lane = threadIdx.x & 31;
  size_t base = (size_t)row * PHID;
  float si = 0.f, se = 0.f;
  #pragma unroll 4
  for (int j = lane; j < PHID; j += 32) {
    float p0 = phi0[base + j], p1 = phi1[base + j], pr = pred[base + j];
    float d = p1 - p0;  si += d * d;
    float e = pr - p1;  se += e * e;
  }
  #pragma unroll
  for (int off = 16; off; off >>= 1) {
    si += __shfl_xor(si, off, 32);
    se += __shfl_xor(se, off, 32);
  }
  if (lane == 0) { impact[row] = __builtin_sqrtf(si); err[row] = se * (1.f / PHID); }
}

// ---------------- KD-tree region lookup on phi_t ----------------
__global__ void kdtree(const float* __restrict__ phi0, const int* __restrict__ sdim,
                       const float* __restrict__ sval, int* __restrict__ rid, int n)
{
  int i = blockIdx.x * blockDim.x + threadIdx.x;
  if (i >= n) return;
  const float* p = phi0 + (size_t)i * PHID;
  int node = 0;
  #pragma unroll
  for (int d = 0; d < KDEPTH; ++d) {
    int   sd = sdim[node];
    float sv = sval[node];
    node = 2 * node + 1 + (p[sd] > sv ? 1 : 0);
  }
  rid[i] = node - NINTD;
}

// ---- order-dependent per-region EMA scan; one thread owns one region ----
// Per-region updates commute across regions, so this is exact and deterministic.
__global__ __launch_bounds__(1024)
void region_scan(const int* __restrict__ rid, const float* __restrict__ err,
                 const float* __restrict__ ema_l, const float* __restrict__ ema_s,
                 const int* __restrict__ cnt0, float* __restrict__ lp, int n)
{
  int r = blockIdx.x * 1024 + threadIdx.x;           // region id 0..4095
  float el = ema_l[r], es = ema_s[r];
  int c = cnt0[r];
  for (int i = 0; i < n; ++i) {
    int ri = rid[i];                                  // uniform -> scalar load
    if (ri == r) {
      float e = err[i];
      if (c == 0) { el = e; es = e; }
      else {
        el = 0.995f * el + 0.005f * e;
        es = 0.9f   * es + 0.1f   * e;
      }
      c++;
      lp[i] = __builtin_fabsf(el - es);
    }
  }
}

// ---------------- deterministic RMS reduction (single block) ----------------
__global__ __launch_bounds__(256)
void reduce_rms(const float* __restrict__ impact, const float* __restrict__ lp,
                const float* __restrict__ rms_imp, const float* __restrict__ rms_lp,
                float* __restrict__ scal, int n)
{
  __shared__ float si[256], sl[256];
  float a = 0.f, b = 0.f;
  for (int i = threadIdx.x; i < n; i += 256) {
    float x = impact[i]; a += x * x;
    float y = lp[i];     b += y * y;
  }
  si[threadIdx.x] = a; sl[threadIdx.x] = b;
  __syncthreads();
  for (int s = 128; s; s >>= 1) {
    if (threadIdx.x < s) {
      si[threadIdx.x] += si[threadIdx.x + s];
      sl[threadIdx.x] += sl[threadIdx.x + s];
    }
    __syncthreads();
  }
  if (threadIdx.x == 0) {
    float mi = si[0] / n, ml = sl[0] / n;
    float ri = rms_imp[0], rl = rms_lp[0];
    scal[0] = __builtin_sqrtf(0.99f * ri * ri + 0.01f * mi);
    scal[1] = __builtin_sqrtf(0.99f * rl * rl + 0.01f * ml);
  }
}

__global__ void bonus_k(const float* __restrict__ impact, const float* __restrict__ lp,
                        const float* __restrict__ scal, float* __restrict__ out, int n)
{
  int i = blockIdx.x * blockDim.x + threadIdx.x;
  if (i >= n) return;
  float rimp = scal[0] + 1e-8f;
  float rlp  = scal[1] + 1e-8f;
  out[i] = impact[i] / rimp + 0.5f * lp[i] / rlp;
}

// ---------------------------------------------------------------------------
extern "C" void kernel_launch(void* const* d_in, const int* in_sizes, int n_in,
                              void* d_out, int out_size, void* d_ws, size_t ws_size,
                              hipStream_t stream) {
  (void)in_sizes; (void)n_in; (void)out_size; (void)ws_size;
  const float* obs      = (const float*)d_in[0];
  const float* next_obs = (const float*)d_in[1];
  const float* W1  = (const float*)d_in[2];
  const float* b1  = (const float*)d_in[3];
  const float* W2  = (const float*)d_in[4];
  const float* b2  = (const float*)d_in[5];
  const float* Wf1 = (const float*)d_in[6];
  const float* bf1 = (const float*)d_in[7];
  const float* Wf2 = (const float*)d_in[8];
  const float* bf2 = (const float*)d_in[9];
  const float* split_val = (const float*)d_in[10];
  const float* ema_long  = (const float*)d_in[11];
  const float* ema_short = (const float*)d_in[12];
  const float* rms_imp   = (const float*)d_in[13];
  const float* rms_lp    = (const float*)d_in[14];
  const int*   actions   = (const int*)d_in[15];
  const int*   split_dim = (const int*)d_in[16];
  const int*   count     = (const int*)d_in[17];
  float* out = (float*)d_out;

  size_t off = 0;
  char* base = (char*)d_ws;
  auto alloc = [&](size_t bytes) -> void* {
    void* p = base + off;
    off += (bytes + 255) & ~(size_t)255;
    return p;
  };
  _Float16* x16   = (_Float16*)alloc((size_t)NTR * OBSD * 2);   // obs/next_obs f16 (reused)
  _Float16* Hbuf  = (_Float16*)alloc((size_t)NTR * HIDD * 2);   // hidden activations (reused)
  float*    phi0  = (float*)   alloc((size_t)NTR * PHID * 4);
  _Float16* phi0h = (_Float16*)alloc((size_t)NTR * PHID * 2);
  float*    phi1  = (float*)   alloc((size_t)NTR * PHID * 4);
  float*    pred  = (float*)   alloc((size_t)NTR * PHID * 4);
  _Float16* W1p   = (_Float16*)alloc((size_t)OBSD * HIDD * 2);
  _Float16* W2p   = (_Float16*)alloc((size_t)HIDD * PHID * 2);
  _Float16* Wf1p  = (_Float16*)alloc((size_t)PHID * HIDD * 2);  // first 512 rows of Wf1
  _Float16* Wf2p  = (_Float16*)alloc((size_t)HIDD * PHID * 2);
  float*    impact= (float*)   alloc((size_t)NTR * 4);
  float*    errv  = (float*)   alloc((size_t)NTR * 4);
  float*    lpv   = (float*)   alloc((size_t)NTR * 4);
  int*      ridv  = (int*)     alloc((size_t)NTR * 4);
  float*    scal  = (float*)   alloc(256);

  // --- convert + pack weights ---
  {
    int n4 = NTR * OBSD / 4;
    cvt_f32_f16<<<n4 / 256, 256, 0, stream>>>(obs, x16, n4);
  }
  pack_w<<<(OBSD * HIDD + 255) / 256, 256, 0, stream>>>(W1,  W1p,  OBSD, HIDD);
  pack_w<<<(HIDD * PHID + 255) / 256, 256, 0, stream>>>(W2,  W2p,  HIDD, PHID);
  pack_w<<<(PHID * HIDD + 255) / 256, 256, 0, stream>>>(Wf1, Wf1p, PHID, HIDD); // rows 0..511
  pack_w<<<(HIDD * PHID + 255) / 256, 256, 0, stream>>>(Wf2, Wf2p, HIDD, PHID);

  dim3 blk(128);
  dim3 gHid(HIDD / 256, NTR / 16);   // N=1024 GEMMs: block tile 16x256
  dim3 gPhi(PHID / 256, NTR / 16);   // N=512  GEMMs

  // encoder(obs): H = relu(obs@W1+b1);  phi_t = H@W2+b2 (f32 + f16)
  gemm_wmma_f16<true,  true,  false, false><<<gHid, blk, 0, stream>>>(
      x16, W1p, b1, Hbuf, (float*)nullptr, NTR, HIDD, OBSD, nullptr, nullptr);
  gemm_wmma_f16<false, true,  true,  false><<<gPhi, blk, 0, stream>>>(
      Hbuf, W2p, b2, phi0h, phi0, NTR, PHID, HIDD, nullptr, nullptr);

  // encoder(next_obs): reuse x16 / Hbuf
  {
    int n4 = NTR * OBSD / 4;
    cvt_f32_f16<<<n4 / 256, 256, 0, stream>>>(next_obs, x16, n4);
  }
  gemm_wmma_f16<true,  true,  false, false><<<gHid, blk, 0, stream>>>(
      x16, W1p, b1, Hbuf, (float*)nullptr, NTR, HIDD, OBSD, nullptr, nullptr);
  gemm_wmma_f16<false, false, true,  false><<<gPhi, blk, 0, stream>>>(
      Hbuf, W2p, b2, (_Float16*)nullptr, phi1, NTR, PHID, HIDD, nullptr, nullptr);

  // forward model: Hf = relu(phi_t@Wf1[:512] + Wf1[512+a] + bf1); pred = Hf@Wf2+bf2
  gemm_wmma_f16<true,  true,  false, true ><<<gHid, blk, 0, stream>>>(
      phi0h, Wf1p, bf1, Hbuf, (float*)nullptr, NTR, HIDD, PHID,
      Wf1 + (size_t)PHID * HIDD, actions);
  gemm_wmma_f16<false, false, true,  false><<<gPhi, blk, 0, stream>>>(
      Hbuf, Wf2p, bf2, (_Float16*)nullptr, pred, NTR, PHID, HIDD, nullptr, nullptr);

  // per-row stats, KD-tree, per-region EMA scan
  rowstats<<<NTR / 8, 256, 0, stream>>>(phi0, phi1, pred, impact, errv);
  kdtree<<<NTR / 256, 256, 0, stream>>>(phi0, split_dim, split_val, ridv, NTR);
  region_scan<<<NREGD / 1024, 1024, 0, stream>>>(ridv, errv, ema_long, ema_short,
                                                 count, lpv, NTR);

  // RMS normalization + bonus
  reduce_rms<<<1, 256, 0, stream>>>(impact, lpv, rms_imp, rms_lp, scal, NTR);
  bonus_k<<<NTR / 256, 256, 0, stream>>>(impact, lpv, scal, out, NTR);
}
